// GATEncoder_87849261072539
// MI455X (gfx1250) — matrix operand order repaired and checked by reading, not compiled, not run on hardware
//
#include <hip/hip_runtime.h>
#include <hip/hip_bf16.h>
#include <stddef.h>

// ---------------------------------------------------------------------------
// GATv2 encoder for MI455X (gfx1250).
// Dense projections: v_wmma_f32_16x16x32_f16, fully branchless tiles
// (rows padded to 16, K padded to 32, weights pre-swizzled to B-fragment
// layout so every fragment load is a global_load_b128).
// Edge pipeline: 3 passes (logits+segmax, exp+segsum, weighted scatter),
// float4-vectorized channel loops, L2-resident gathers + f32 atomics.
// ---------------------------------------------------------------------------

typedef __attribute__((ext_vector_type(16))) _Float16 v16h;
typedef __attribute__((ext_vector_type(8)))  float    v8f;

union Frag16 { v16h v; unsigned u[8]; uint4 q[2]; };

__device__ __forceinline__ unsigned ford(float f) {
    unsigned u = __float_as_uint(f);
    return (u & 0x80000000u) ? ~u : (u | 0x80000000u);
}
__device__ __forceinline__ float fdeord(unsigned u) {
    return __uint_as_float((u & 0x80000000u) ? (u & 0x7FFFFFFFu) : ~u);
}
__device__ __forceinline__ unsigned short f2h_bits(float f) {
    _Float16 h = (_Float16)f;
    return __builtin_bit_cast(unsigned short, h);
}

// ---------------- utility kernels ----------------

__global__ void zero_u32_kernel(unsigned* __restrict__ p, long long n) {
    long long i = (long long)blockIdx.x * blockDim.x + threadIdx.x;
    if (i < n) p[i] = 0u;
}

// in[rows,Kin] (f32, dense) -> out[rows,Kpad] (f16 bits); pad cols pre-zeroed.
__global__ void f32_to_f16_pad_kernel(const float* __restrict__ in,
                                      unsigned short* __restrict__ out,
                                      int rows, int Kin, int Kpad) {
    int i = blockIdx.x * blockDim.x + threadIdx.x;
    if (i >= rows * Kin) return;
    int r = i / Kin, c = i % Kin;
    out[(size_t)r * Kpad + c] = f2h_bits(in[i]);
}

// Repack W[Kreal,DOUT] (f32) into the WMMA B-fragment stream (f16 pairs):
// ws[(((kt*(DOUT/16))+tn)*32 + lane)*8 + v] packs W[k,col],W[k+1,col]
// with col = tn*16+(lane&15), k = kt*32+(lane>>4)*16+2v. Zero-pad k>=Kreal.
__global__ void repack_w_kernel(const float* __restrict__ W,
                                unsigned* __restrict__ ws,
                                int Kreal, int Kpad, int DOUT) {
    int idx = blockIdx.x * blockDim.x + threadIdx.x;
    int total = (Kpad / 32) * (DOUT / 16) * 32 * 8;
    if (idx >= total) return;
    int v = idx & 7;
    int lane = (idx >> 3) & 31;
    int t = idx >> 8;
    int tn = t % (DOUT / 16);
    int kt = t / (DOUT / 16);
    int col = tn * 16 + (lane & 15);
    int k = kt * 32 + (lane >> 4) * 16 + 2 * v;
    unsigned lo = 0, hi = 0;
    if (k < Kreal)     lo = f2h_bits(W[(size_t)k * DOUT + col]);
    if (k + 1 < Kreal) hi = f2h_bits(W[(size_t)(k + 1) * DOUT + col]);
    ws[idx] = lo | (hi << 16);
}

__global__ void deg_loop_accum_kernel(const int* __restrict__ dst,
                                      const float* __restrict__ eattr,
                                      float* __restrict__ deg,
                                      float* __restrict__ lsum, int E) {
    int i = blockIdx.x * blockDim.x + threadIdx.x;
    if (i >= E) return;
    int d = dst[i];
    atomicAdd(&deg[d], 1.0f);
#pragma unroll
    for (int k = 0; k < 8; ++k)
        atomicAdd(&lsum[(size_t)d * 8 + k], eattr[(size_t)i * 8 + k]);
}

__global__ void loop_norm_kernel(float* __restrict__ lsum,
                                 const float* __restrict__ deg, int n8) {
    int i = blockIdx.x * blockDim.x + threadIdx.x;
    if (i >= n8) return;
    float dg = deg[i >> 3];
    if (dg < 1.0f) dg = 1.0f;
    lsum[i] /= dg;
}

// ---------------- WMMA dense projection (branchless) ----------------
// Y[Npad, DOUT] = X[Npad, K] @ W + bias. K multiple of 32, rows padded.
// One wave computes one 16x16 tile; all fragment loads are b128.

template <int K, int DOUT>
__global__ void gemm_f16_wmma_kernel(const unsigned short* __restrict__ X,
                                     const unsigned* __restrict__ Wswz,
                                     const float* __restrict__ bias,
                                     float* __restrict__ Y, int numTiles) {
    const int lane = threadIdx.x & 31;
    const int wave = threadIdx.x >> 5;
    const int tile = blockIdx.x * (blockDim.x >> 5) + wave;
    if (tile >= numTiles) return;
    constexpr int tilesN = DOUT / 16;
    const int tm = tile / tilesN;
    const int tn = tile % tilesN;
    const int row = tm * 16 + (lane & 15);
    const int col = tn * 16 + (lane & 15);
    const int half = lane >> 4;

    v8f acc;
    float bv = bias ? bias[col] : 0.0f;
#pragma unroll
    for (int r = 0; r < 8; ++r) acc[r] = bv;

#pragma unroll
    for (int kt = 0; kt < K / 32; ++kt) {
        Frag16 a, b;
        const uint4* pa =
            (const uint4*)(X + (size_t)row * K + kt * 32 + 16 * half);
        a.q[0] = pa[0];               // halves [k0+16h  .. k0+16h+7]
        a.q[1] = pa[1];               // halves [k0+16h+8.. k0+16h+15]
        const uint4* pb =
            (const uint4*)(Wswz + ((size_t)(kt * tilesN + tn) * 32 + lane) * 8);
        b.q[0] = pb[0];
        b.q[1] = pb[1];
        acc = __builtin_amdgcn_wmma_f32_16x16x32_f16(
            false, a.v, false, b.v, (short)0, acc, false, false);
    }

    float* yb = Y + (size_t)(tm * 16) * DOUT + col;
#pragma unroll
    for (int r = 0; r < 8; ++r)
        yb[(size_t)(r + half * 8) * DOUT] = acc[r];
}

// ---------------- edge pass A: logits + segment max ----------------
// Edge i >= E is the self-loop of node (i - E) with attr = loop_attr.

template <int D, int H>
__global__ void edge_logits_kernel(const int* __restrict__ src,
                                   const int* __restrict__ dst,
                                   const float* __restrict__ eattr,
                                   const float* __restrict__ loop_attr,
                                   const float* __restrict__ xl,
                                   const float* __restrict__ xr,
                                   const float* __restrict__ We,
                                   const float* __restrict__ att,
                                   float* __restrict__ logits,
                                   unsigned* __restrict__ mseg,
                                   int E, int Nn) {
    __shared__ float sWe[8 * D];
    __shared__ float sAtt[D];
    for (int t = threadIdx.x; t < 8 * D; t += blockDim.x) sWe[t] = We[t];
    for (int t = threadIdx.x; t < D; t += blockDim.x) sAtt[t] = att[t];
    __syncthreads();

    int i = blockIdx.x * blockDim.x + threadIdx.x;
    if (i >= E + Nn) return;
    int s, d;
    const float* ea;
    if (i < E) { s = src[i]; d = dst[i]; ea = eattr + (size_t)i * 8; }
    else       { s = d = i - E;          ea = loop_attr + (size_t)(i - E) * 8; }
    float a[8];
#pragma unroll
    for (int k = 0; k < 8; ++k) a[k] = ea[k];

    constexpr int C = D / H;
    const float* xls = xl + (size_t)s * D;
    const float* xrd = xr + (size_t)d * D;
#pragma unroll
    for (int h = 0; h < H; ++h) {
        float lg = 0.0f;
        for (int c4 = 0; c4 < C / 4; ++c4) {
            int j = h * C + c4 * 4;
            float4 va = *(const float4*)(xls + j);
            float4 vb = *(const float4*)(xrd + j);
            const float* pa = &va.x;
            const float* pb = &vb.x;
#pragma unroll
            for (int t = 0; t < 4; ++t) {
                float e = 0.0f;
#pragma unroll
                for (int k = 0; k < 8; ++k) e += a[k] * sWe[k * D + j + t];
                float m = pa[t] + pb[t] + e;
                m = (m > 0.0f) ? m : 0.2f * m;   // leaky relu
                lg += m * sAtt[j + t];
            }
        }
        logits[(size_t)i * H + h] = lg;
        atomicMax(&mseg[(size_t)d * H + h], ford(lg));
    }
}

// ---------------- edge pass B: exp + segment sum ----------------

template <int H>
__global__ void edge_expsum_kernel(const int* __restrict__ dst,
                                   const float* __restrict__ logits,
                                   const unsigned* __restrict__ mseg,
                                   float* __restrict__ expv,
                                   float* __restrict__ sseg, int E, int Nn) {
    long long idx = (long long)blockIdx.x * blockDim.x + threadIdx.x;
    if (idx >= (long long)(E + Nn) * H) return;
    int i = (int)(idx / H);
    int h = (int)(idx % H);
    int d = (i < E) ? dst[i] : (i - E);
    float m = fdeord(mseg[(size_t)d * H + h]);
    float ev = __expf(logits[idx] - m);
    expv[idx] = ev;
    atomicAdd(&sseg[(size_t)d * H + h], ev);
}

// ---------------- edge pass C: weighted scatter-add ----------------

template <int D, int H>
__global__ void edge_aggregate_kernel(const int* __restrict__ src,
                                      const int* __restrict__ dst,
                                      const float* __restrict__ xl,
                                      const float* __restrict__ expv,
                                      const float* __restrict__ sseg,
                                      float* __restrict__ agg, int E, int Nn) {
    int i = blockIdx.x * blockDim.x + threadIdx.x;
    if (i >= E + Nn) return;
    int s, d;
    if (i < E) { s = src[i]; d = dst[i]; } else { s = d = i - E; }
    constexpr int C = D / H;
    const float* xls = xl + (size_t)s * D;
    float* ad = agg + (size_t)d * D;
#pragma unroll
    for (int h = 0; h < H; ++h) {
        float w = expv[(size_t)i * H + h] / sseg[(size_t)d * H + h];
        for (int c4 = 0; c4 < C / 4; ++c4) {
            int j = h * C + c4 * 4;
            float4 v = *(const float4*)(xls + j);
            atomicAdd(&ad[j + 0], w * v.x);
            atomicAdd(&ad[j + 1], w * v.y);
            atomicAdd(&ad[j + 2], w * v.z);
            atomicAdd(&ad[j + 3], w * v.w);
        }
    }
}

// ---------------- epilogues ----------------

__global__ void epilogue_relu_f16_kernel(const float* __restrict__ agg,
                                         const float* __restrict__ res,
                                         const float* __restrict__ bias,
                                         unsigned short* __restrict__ hf16,
                                         int total, int Dd) {
    int idx = blockIdx.x * blockDim.x + threadIdx.x;
    if (idx >= total) return;
    int j = idx % Dd;
    float v = agg[idx] + bias[j];
    if (res) v += res[idx];
    v = (v > 0.0f) ? v : 0.0f;
    hf16[idx] = f2h_bits(v);
}

__global__ void add_bias_kernel(float* __restrict__ out,
                                const float* __restrict__ bias,
                                int total, int Dd) {
    int idx = blockIdx.x * blockDim.x + threadIdx.x;
    if (idx >= total) return;
    out[idx] += bias[idx % Dd];
}

// ---------------- host launch ----------------

static inline int cdiv_i(long long a, long long b) { return (int)((a + b - 1) / b); }

extern "C" void kernel_launch(void* const* d_in, const int* in_sizes, int n_in,
                              void* d_out, int out_size, void* d_ws, size_t ws_size,
                              hipStream_t stream) {
    const float* x    = (const float*)d_in[0];
    const int*  eidx  = (const int*)d_in[1];
    const float* eat  = (const float*)d_in[2];
    const float* Wl1  = (const float*)d_in[3];  const float* bl1 = (const float*)d_in[4];
    const float* Wr1  = (const float*)d_in[5];  const float* br1 = (const float*)d_in[6];
    const float* We1  = (const float*)d_in[7];  const float* att1 = (const float*)d_in[8];
    const float* b1   = (const float*)d_in[9];
    const float* Wl2  = (const float*)d_in[10]; const float* bl2 = (const float*)d_in[11];
    const float* Wr2  = (const float*)d_in[12]; const float* br2 = (const float*)d_in[13];
    const float* We2  = (const float*)d_in[14]; const float* att2 = (const float*)d_in[15];
    const float* b2   = (const float*)d_in[16]; const float* Rw2 = (const float*)d_in[17];
    const float* Wl3  = (const float*)d_in[18]; const float* bl3 = (const float*)d_in[19];
    const float* Wr3  = (const float*)d_in[20]; const float* br3 = (const float*)d_in[21];
    const float* We3  = (const float*)d_in[22]; const float* att3 = (const float*)d_in[23];
    const float* b3   = (const float*)d_in[24];

    const int Nn = in_sizes[0] / 16;
    const int E  = in_sizes[1] / 2;
    const int E2 = E + Nn;
    const int Npad = (Nn + 15) & ~15;
    const int* src = eidx;
    const int* dst = eidx + E;
    float* out = (float*)d_out;

    // bump allocator over d_ws
    char* wsB = (char*)d_ws;
    size_t off = 0;
    auto alloc = [&](size_t bytes) -> void* {
        void* p = wsB + off;
        off = (off + bytes + 255) & ~(size_t)255;
        return p;
    };
    float* deg       = (float*)alloc(sizeof(float) * Nn);
    float* loop_attr = (float*)alloc(sizeof(float) * (size_t)Nn * 8);
    unsigned short* xh  = (unsigned short*)alloc(2ull * Npad * 32);   // K padded 16->32
    unsigned short* h1h = (unsigned short*)alloc(2ull * Npad * 32);
    unsigned short* h2h = (unsigned short*)alloc(2ull * Npad * 128);
    unsigned* wl1s = (unsigned*)alloc(4ull * 512);   // (32/32)*(32/16)*256
    unsigned* wr1s = (unsigned*)alloc(4ull * 512);
    unsigned* wl2s = (unsigned*)alloc(4ull * 2048);  // (32/32)*(128/16)*256
    unsigned* wr2s = (unsigned*)alloc(4ull * 2048);
    unsigned* rw2s = (unsigned*)alloc(4ull * 2048);
    unsigned* wl3s = (unsigned*)alloc(4ull * 2048);  // (128/32)*(32/16)*256
    unsigned* wr3s = (unsigned*)alloc(4ull * 2048);
    float* xl     = (float*)alloc(sizeof(float) * (size_t)Npad * 128);
    float* xr     = (float*)alloc(sizeof(float) * (size_t)Npad * 128);
    float* agg    = (float*)alloc(sizeof(float) * (size_t)Nn * 128);
    float* r2     = (float*)alloc(sizeof(float) * (size_t)Npad * 128);
    float* logits = (float*)alloc(sizeof(float) * (size_t)E2 * 4);
    float* expv   = (float*)alloc(sizeof(float) * (size_t)E2 * 4);
    unsigned* mseg = (unsigned*)alloc(sizeof(unsigned) * (size_t)Nn * 4);
    float* sseg    = (float*)alloc(sizeof(float) * (size_t)Nn * 4);
    (void)ws_size; (void)n_in; (void)out_size;

    const int B = 256;
    auto zero = [&](void* p, long long ndw) {
        zero_u32_kernel<<<cdiv_i(ndw, B), B, 0, stream>>>((unsigned*)p, ndw);
    };

    // ---- self-loop attr (mean of incoming edge_attr) ----
    zero(deg, Nn);
    zero(loop_attr, (long long)Nn * 8);
    deg_loop_accum_kernel<<<cdiv_i(E, B), B, 0, stream>>>(dst, eat, deg, loop_attr, E);
    loop_norm_kernel<<<cdiv_i((long long)Nn * 8, B), B, 0, stream>>>(loop_attr, deg, Nn * 8);

    // ---- f16 activation buffers (zero pads) + input conversion ----
    zero(xh,  (long long)Npad * 16);   // dwords
    zero(h1h, (long long)Npad * 16);
    zero(h2h, (long long)Npad * 64);
    f32_to_f16_pad_kernel<<<cdiv_i((long long)Nn * 16, B), B, 0, stream>>>(x, xh, Nn, 16, 32);

    // ---- weight repack into B-fragment layout ----
    repack_w_kernel<<<cdiv_i(512, B), B, 0, stream>>>(Wl1, wl1s, 16, 32, 32);
    repack_w_kernel<<<cdiv_i(512, B), B, 0, stream>>>(Wr1, wr1s, 16, 32, 32);
    repack_w_kernel<<<cdiv_i(2048, B), B, 0, stream>>>(Wl2, wl2s, 32, 32, 128);
    repack_w_kernel<<<cdiv_i(2048, B), B, 0, stream>>>(Wr2, wr2s, 32, 32, 128);
    repack_w_kernel<<<cdiv_i(2048, B), B, 0, stream>>>(Rw2, rw2s, 32, 32, 128);
    repack_w_kernel<<<cdiv_i(2048, B), B, 0, stream>>>(Wl3, wl3s, 128, 128, 32);
    repack_w_kernel<<<cdiv_i(2048, B), B, 0, stream>>>(Wr3, wr3s, 128, 128, 32);

    const int tilesM = Npad / 16;

    // ================= layer 1: in 16(->32) -> 4 heads x 8 (D=32) =================
    {
        int tiles = tilesM * 2;
        int blk = cdiv_i(tiles, 8);
        gemm_f16_wmma_kernel<32, 32><<<blk, B, 0, stream>>>(xh, wl1s, bl1, xl, tiles);
        gemm_f16_wmma_kernel<32, 32><<<blk, B, 0, stream>>>(xh, wr1s, br1, xr, tiles);
        zero(mseg, (long long)Nn * 4);
        zero(sseg, (long long)Nn * 4);
        edge_logits_kernel<32, 4><<<cdiv_i(E2, B), B, 0, stream>>>(
            src, dst, eat, loop_attr, xl, xr, We1, att1, logits, mseg, E, Nn);
        edge_expsum_kernel<4><<<cdiv_i((long long)E2 * 4, B), B, 0, stream>>>(
            dst, logits, mseg, expv, sseg, E, Nn);
        zero(agg, (long long)Nn * 32);
        edge_aggregate_kernel<32, 4><<<cdiv_i(E2, B), B, 0, stream>>>(
            src, dst, xl, expv, sseg, agg, E, Nn);
        epilogue_relu_f16_kernel<<<cdiv_i((long long)Nn * 32, B), B, 0, stream>>>(
            agg, nullptr, b1, h1h, Nn * 32, 32);
    }

    // ============ layer 2: in 32 -> 4 heads x 32 (D=128) + residual ============
    {
        int tiles = tilesM * 8;
        int blk = cdiv_i(tiles, 8);
        gemm_f16_wmma_kernel<32, 128><<<blk, B, 0, stream>>>(h1h, wl2s, bl2, xl, tiles);
        gemm_f16_wmma_kernel<32, 128><<<blk, B, 0, stream>>>(h1h, wr2s, br2, xr, tiles);
        gemm_f16_wmma_kernel<32, 128><<<blk, B, 0, stream>>>(h1h, rw2s, nullptr, r2, tiles);
        zero(mseg, (long long)Nn * 4);
        zero(sseg, (long long)Nn * 4);
        edge_logits_kernel<128, 4><<<cdiv_i(E2, B), B, 0, stream>>>(
            src, dst, eat, loop_attr, xl, xr, We2, att2, logits, mseg, E, Nn);
        edge_expsum_kernel<4><<<cdiv_i((long long)E2 * 4, B), B, 0, stream>>>(
            dst, logits, mseg, expv, sseg, E, Nn);
        zero(agg, (long long)Nn * 128);
        edge_aggregate_kernel<128, 4><<<cdiv_i(E2, B), B, 0, stream>>>(
            src, dst, xl, expv, sseg, agg, E, Nn);
        epilogue_relu_f16_kernel<<<cdiv_i((long long)Nn * 128, B), B, 0, stream>>>(
            agg, r2, b2, h2h, Nn * 128, 128);
    }

    // ================= layer 3: in 128 -> 1 head x 32 (D=32) =================
    {
        int tiles = tilesM * 2;
        int blk = cdiv_i(tiles, 8);
        gemm_f16_wmma_kernel<128, 32><<<blk, B, 0, stream>>>(h2h, wl3s, bl3, xl, tiles);
        gemm_f16_wmma_kernel<128, 32><<<blk, B, 0, stream>>>(h2h, wr3s, br3, xr, tiles);
        zero(mseg, Nn);
        zero(sseg, Nn);
        edge_logits_kernel<32, 1><<<cdiv_i(E2, B), B, 0, stream>>>(
            src, dst, eat, loop_attr, xl, xr, We3, att3, logits, mseg, E, Nn);
        edge_expsum_kernel<1><<<cdiv_i(E2, B), B, 0, stream>>>(
            dst, logits, mseg, expv, sseg, E, Nn);
        zero(out, (long long)Nn * 32);
        edge_aggregate_kernel<32, 1><<<cdiv_i(E2, B), B, 0, stream>>>(
            src, dst, xl, expv, sseg, out, E, Nn);
        add_bias_kernel<<<cdiv_i((long long)Nn * 32, B), B, 0, stream>>>(out, b3, Nn * 32, 32);
    }
}
